// MNIST_Type1_Template_Unroll_74217034875443
// MI455X (gfx1250) — matrix operand order repaired and checked by reading, not compiled, not run on hardware
//
#include <hip/hip_runtime.h>

// ---------------------------------------------------------------------------
// Fused locally-connected MNIST head for MI455X (gfx1250, wave32, WMMA).
//   conv1+conv2 (VALU, streaming) -> FC1 (v_wmma_f32_16x16x32_f16) -> ReLU
//   -> FC2 (WMMA, N padded 10->16) -> f32 logits.
// B-matrix (fc1 weights) is staged global->LDS with ASYNC copies
// (global_load_async_to_lds_b128, depth-2 pipeline, 3 slab buffers) and
// shared by all 8 waves: fc1 is read from L2 once per 128-row block
// (~310 MB total vs 2.5 GB with per-wave loads). 320 KB LDS, 256 thr/block.
// ---------------------------------------------------------------------------

typedef __attribute__((ext_vector_type(16))) _Float16 v16h;
typedef __attribute__((ext_vector_type(8)))  _Float16 v8h;
typedef __attribute__((ext_vector_type(8)))  float    v8f;

#define KPAD    1184          // 1176 padded to 37 * 32
#define KSTEPS  37
#define ROWS    128           // batch rows per block
#define A_HALFS (ROWS * KPAD) // 151552 halfs = 303104 B
#define SLAB_H  4096          // 32k x 128n halfs = 8192 B per slab
#define NBUF    3             // depth-2 async pipeline

__device__ __forceinline__ v16h cat8(v8h lo, v8h hi) {
  v16h r;
#pragma unroll
  for (int i = 0; i < 8; ++i) { r[i] = lo[i]; r[i + 8] = hi[i]; }
  return r;
}

__device__ __forceinline__ v8f vzero8f() {
  v8f r;
#pragma unroll
  for (int i = 0; i < 8; ++i) r[i] = 0.f;
  return r;
}

// Async copy 16B global -> LDS (ASYNCcnt-tracked, lands directly in LDS).
__device__ __forceinline__ void async_g2l_b128(unsigned int lds_byte_off,
                                               const _Float16* g) {
  asm volatile("global_load_async_to_lds_b128 %0, %1, off"
               :: "v"(lds_byte_off), "v"(g) : "memory");
}

// Each wave's 2 instructions cover columns [16*wave, 16*wave+16) of the slab.
__device__ __forceinline__ void prefetch_slab(const _Float16* __restrict__ fc1h,
                                              unsigned int slab_off, int nc,
                                              int ks, int wave, int lane) {
#pragma unroll
  for (int j = 0; j < 2; ++j) {
    const int seg = wave * 64 + j * 32 + lane;   // 0..511, 16B each
    const int c   = seg >> 2;                    // column 0..127
    const int kk  = (seg & 3) * 8;               // k-offset in halfs
    const _Float16* g = fc1h + (long)(nc * 128 + c) * KPAD + ks * 32 + kk;
    async_g2l_b128(slab_off + (unsigned int)seg * 16, g);
  }
}

// --------------------------- weight prep (f32 -> f16, padded) ---------------
__global__ void prep_weights(const float* __restrict__ fc1,
                             const float* __restrict__ fc2,
                             _Float16* __restrict__ fc1h,
                             _Float16* __restrict__ fc2h) {
  long i = (long)blockIdx.x * blockDim.x + threadIdx.x;
  const long N1 = 512L * KPAD;
  if (i < N1) {
    int n = (int)(i / KPAD);
    int k = (int)(i % KPAD);
    float v = (k < 1176) ? fc1[(long)n * 1176 + k] : 0.f;
    fc1h[i] = (_Float16)v;
  } else {
    long j = i - N1;                     // [16][512] padded fc2
    if (j < 16L * 512) {
      int n = (int)(j / 512);
      int k = (int)(j % 512);
      float v = (n < 10) ? fc2[(long)n * 512 + k] : 0.f;
      fc2h[j] = (_Float16)v;
    }
  }
}

// --------------------------- fused network ----------------------------------
__global__ __launch_bounds__(256)
void fused_mnist(const float* __restrict__ x,
                 const float* __restrict__ c1w,     // [3][196][1][2][2]
                 const float* __restrict__ c2w,     // [24][49][3][2][2]
                 const _Float16* __restrict__ fc1h, // [512][KPAD], n-major
                 const _Float16* __restrict__ fc2h, // [16][512],  n-major
                 float* __restrict__ out) {         // [B][10]
  extern __shared__ _Float16 smem[];
  _Float16* Ash = smem;                  // [ROWS][KPAD] out2 (f16)
  _Float16* Bsh = smem + A_HALFS;        // 3 x [128 cols][32 k] B slabs
  // LDS byte offset of Bsh for the async engine (generic addr low 32 bits).
  const unsigned int BshOff = (unsigned int)(unsigned long long)(void*)Bsh;

  const int tid  = threadIdx.x;
  const int lane = tid & 31;
  const int wave = tid >> 5;             // 0..7
  const int nloc = lane & 15;
  const int half = lane >> 4;

  // ============== conv phase: 2 threads per sample, streaming ==============
  {
    const int s   = tid >> 1;            // local sample 0..127
    const int seg = tid & 1;
    const long gb = (long)blockIdx.x * ROWS + s;
    const float* xs = x + gb * 784;
    const int p0 = seg ? 25 : 0;
    const int p1 = seg ? 49 : 25;
    for (int pos = p0; pos < p1; ++pos) {
      const int h2 = pos / 7, w2 = pos % 7;
      float xp[16];                       // aligned 4x4 input patch
#pragma unroll
      for (int r = 0; r < 4; ++r) {
        const float4 v = *(const float4*)(xs + (4 * h2 + r) * 28 + 4 * w2);
        xp[r * 4 + 0] = v.x; xp[r * 4 + 1] = v.y;
        xp[r * 4 + 2] = v.z; xp[r * 4 + 3] = v.w;
      }
      float o1[12];                       // layer-1 relu'd values
#pragma unroll
      for (int k = 0; k < 3; ++k) {
#pragma unroll
        for (int i1 = 0; i1 < 2; ++i1) {
#pragma unroll
          for (int j1 = 0; j1 < 2; ++j1) {
            const int H1 = 2 * h2 + i1, W1 = 2 * w2 + j1;
            const float* wp = c1w + k * 784 + (H1 * 14 + W1) * 4;
            float a = xp[(2 * i1) * 4 + 2 * j1]         * wp[0]
                    + xp[(2 * i1) * 4 + 2 * j1 + 1]     * wp[1]
                    + xp[(2 * i1 + 1) * 4 + 2 * j1]     * wp[2]
                    + xp[(2 * i1 + 1) * 4 + 2 * j1 + 1] * wp[3];
            o1[k * 4 + i1 * 2 + j1] = a > 0.f ? a : 0.f;
          }
        }
      }
#pragma unroll
      for (int c = 0; c < 24; ++c) {      // layer 2 -> A row (f16)
        const float* wp2 = c2w + c * 588 + pos * 12;
        float a = 0.f;
#pragma unroll
        for (int q = 0; q < 12; ++q) a += o1[q] * wp2[q];
        Ash[(long)s * KPAD + c * 49 + pos] = (_Float16)(a > 0.f ? a : 0.f);
      }
    }
    if (tid < ROWS) {                     // zero K padding 1176..1183
#pragma unroll
      for (int k = 1176; k < KPAD; ++k) Ash[(long)tid * KPAD + k] = (_Float16)0.f;
    }
  }
  __syncthreads();

  // ============== GEMM phase: wave w owns M-tile rows [16w, 16w+16) ========
  const _Float16* Abase = Ash + (long)(wave * 16) * KPAD;
  v8f acc2 = vzero8f();                   // FC2 16x16 output tile

#pragma unroll 1
  for (int nc = 0; nc < 4; ++nc) {        // 4 x 128 columns of FC1
    v8f acc[8];
#pragma unroll
    for (int t = 0; t < 8; ++t) acc[t] = vzero8f();

    // prime the async pipeline: slabs 0 and 1 in flight, drain slab 0
    prefetch_slab(fc1h, BshOff + 0 * 8192, nc, 0, wave, lane);
    prefetch_slab(fc1h, BshOff + 1 * 8192, nc, 1, wave, lane);
    asm volatile("s_wait_asynccnt 2" ::: "memory");
    __syncthreads();

#pragma unroll 1
    for (int ks = 0; ks < KSTEPS; ++ks) {
      if (ks + 2 < KSTEPS)
        prefetch_slab(fc1h, BshOff + ((ks + 2) % NBUF) * 8192, nc, ks + 2, wave, lane);

      // A fragment (16-bit 16x32 layout): 2 x ds_load_b128
      const _Float16* ap = Abase + (long)nloc * KPAD + ks * 32 + 8 * half;
      v8h alo = *(const v8h*)ap;
      v8h ahi = *(const v8h*)(ap + 16);
      v16h a = cat8(alo, ahi);

      const _Float16* bcur = Bsh + (ks % NBUF) * SLAB_H;  // [128 cols][32 k]
#pragma unroll
      for (int t = 0; t < 8; ++t) {
        const _Float16* bp = bcur + (t * 16 + nloc) * 32 + half * 8;
        v8h blo = *(const v8h*)bp;
        v8h bhi = *(const v8h*)(bp + 16);
        v16h b = cat8(blo, bhi);
        acc[t] = __builtin_amdgcn_wmma_f32_16x16x32_f16(
            false, a, false, b, (short)0, acc[t], false, false);
      }

      if (ks + 1 < KSTEPS) {              // make slab ks+1 visible to all
        if (ks + 2 < KSTEPS) { asm volatile("s_wait_asynccnt 2" ::: "memory"); }
        else                 { asm volatile("s_wait_asynccnt 0" ::: "memory"); }
        __syncthreads();
      }
    }
    __syncthreads();                      // all waves done reading B slabs

    // FC2: route h through per-wave 16x32 staging (reuses slab buffer 0 --
    // wave w's async-prefetch region == wave w's staging bytes, so private).
    _Float16* stag = Bsh + wave * 512;    // 1 KB per wave inside slab 0
#pragma unroll
    for (int k2 = 0; k2 < 4; ++k2) {
#pragma unroll
      for (int tt = 0; tt < 2; ++tt) {
        const int t = 2 * k2 + tt;
#pragma unroll
        for (int r = 0; r < 8; ++r) {
          float v = acc[t][r];
          v = v > 0.f ? v : 0.f;          // ReLU
          stag[(r + 8 * half) * 32 + tt * 16 + nloc] = (_Float16)v;
        }
      }
      asm volatile("s_wait_dscnt 0" ::: "memory");  // same-wave LDS RAW
      const _Float16* ap2 = stag + nloc * 32 + 16 * half;
      v8h alo2 = *(const v8h*)ap2;
      v8h ahi2 = *(const v8h*)(ap2 + 8);
      v16h a2 = cat8(alo2, ahi2);
      const v16h b2 = *(const v16h*)(fc2h + (long)nloc * 512 + nc * 128 + k2 * 32 + half * 16);
      acc2 = __builtin_amdgcn_wmma_f32_16x16x32_f16(
          false, a2, false, b2, (short)0, acc2, false, false);
      asm volatile("" ::: "memory");
    }
    __syncthreads();                      // staging quiesced before next nc
  }

  // ============== write logits (f32), columns n < 10 =======================
  const long row0 = (long)blockIdx.x * ROWS + wave * 16;
  if (nloc < 10) {
#pragma unroll
    for (int r = 0; r < 8; ++r) {
      const long b = row0 + r + 8 * half; // C-layout row mapping
      out[b * 10 + nloc] = acc2[r];
    }
  }
}

// --------------------------- host launcher ----------------------------------
extern "C" void kernel_launch(void* const* d_in, const int* in_sizes, int n_in,
                              void* d_out, int out_size, void* d_ws, size_t ws_size,
                              hipStream_t stream) {
  const float* x    = (const float*)d_in[0];
  const float* c1w  = (const float*)d_in[1];
  const float* c2w  = (const float*)d_in[2];
  const float* fc1  = (const float*)d_in[3];
  const float* fc2  = (const float*)d_in[4];
  float* out        = (float*)d_out;

  // workspace: fc1h [512][KPAD] f16 (1,212,416 B) + fc2h [16][512] f16 (16,384 B)
  _Float16* fc1h = (_Float16*)d_ws;
  _Float16* fc2h = fc1h + 512L * KPAD;

  const int B = in_sizes[0] / 784;        // 32768

  {
    const long total = 512L * KPAD + 16L * 512;   // 614,400 elements
    const int threads = 256;
    const int blocks = (int)((total + threads - 1) / threads);
    prep_weights<<<blocks, threads, 0, stream>>>(fc1, fc2, fc1h, fc2h);
  }
  {
    // 303,104 B (A) + 3 * 8,192 B (B slabs) = 327,680 B = 320 KB (WGP max)
    const size_t shmem = ((size_t)A_HALFS + NBUF * SLAB_H) * sizeof(_Float16);
    fused_mnist<<<B / ROWS, 256, shmem, stream>>>(x, c1w, c2w, fc1h, fc2h, out);
  }
}
